// ProductQuantizer_82695300317335
// MI455X (gfx1250) — matrix operand order repaired and checked by reading, not compiled, not run on hardware
//
#include <hip/hip_runtime.h>

typedef __attribute__((ext_vector_type(16))) _Float16 v16h;
typedef __attribute__((ext_vector_type(8)))  float    v8f;

#define Gq     4
#define Kq     1024
#define Dq     64
#define NTOK   (8 * 4096)
#define TT     4            // token tiles (of 16) per wave -> 64 tokens/wave
#define DECAYf 0.99f
#define OMDf   0.01f
#define EPSf   1e-5f

// ---------------------------------------------------------------------------
// Kernel 0: embed f32->f16 (WMMA B layout source), h2 = 0.5*||e||^2,
// zero count / embed_sum accumulators (fresh every launch).
// ---------------------------------------------------------------------------
__global__ void pq_prep_kernel(const float* __restrict__ embed,
                               _Float16* __restrict__ embed_f16,
                               float* __restrict__ h2,
                               float* __restrict__ counts,
                               float* __restrict__ embed_sum) {
    int gk = blockIdx.x * blockDim.x + threadIdx.x;
    if (gk >= Gq * Kq) return;
    const float* er = embed + (size_t)gk * Dq;
    _Float16*    eh = embed_f16 + (size_t)gk * Dq;
    float*       es = embed_sum + (size_t)gk * Dq;
    float acc = 0.0f;
    #pragma unroll 4
    for (int i = 0; i < Dq; ++i) {
        float v = er[i];
        acc += v * v;
        eh[i] = (_Float16)v;
        es[i] = 0.0f;
    }
    h2[gk] = 0.5f * acc;
    counts[gk] = 0.0f;
}

// ---------------------------------------------------------------------------
// Kernel 1: nearest-code assignment via WMMA.
// Block = 256 threads = 8 waves. wave -> (group g = wave&3, 64-token strip).
// Each wave holds 4 A tiles (64 tokens x 64 dims, f16) in registers and
// streams the group's codebook once: per 16-code tile, 2 B fragments feed
// 8 v_wmma_f32_16x16x32_f16 (4 independent accumulators -> no RAW NOPs).
// score = dot - 0.5*||e||^2 ; per-lane running argmax; 16-lane shuffle finish.
// ---------------------------------------------------------------------------
__global__ void __launch_bounds__(256)
pq_assign_kernel(const float* __restrict__ x,
                 const _Float16* __restrict__ embed_f16,
                 const float* __restrict__ h2,
                 int* __restrict__ ind_out) {
    const int lane = threadIdx.x & 31;
    const int wave = threadIdx.x >> 5;
    const int g    = wave & 3;
    const int sub  = wave >> 2;
    const int tok0 = blockIdx.x * (2 * TT * 16) + sub * (TT * 16);
    const int lh   = lane & 15;
    const bool hi  = (lane >= 16);
    const int kb   = hi ? 8 : 0;

    // --- A fragments: contiguous k order (k = 16*(lane/16) + s), same as B.
    // lanes 0-15: M=lane,  K = {0..15}; lanes 16-31: M=lane-16, K = {16..31}
    v16h a0[TT], a1[TT];
    #pragma unroll
    for (int t = 0; t < TT; ++t) {
        const float* xrow = x + ((size_t)(tok0 + t * 16 + lh) * Gq + g) * Dq;
        #pragma unroll
        for (int i = 0; i < 8; ++i) {
            a0[t][i]     = (_Float16)xrow[2 * kb + i];
            a0[t][8 + i] = (_Float16)xrow[2 * kb + 8 + i];
            a1[t][i]     = (_Float16)xrow[32 + 2 * kb + i];
            a1[t][8 + i] = (_Float16)xrow[32 + 2 * kb + 8 + i];
        }
    }

    float rmax[TT][8];
    int   ridx[TT][8];
    #pragma unroll
    for (int t = 0; t < TT; ++t)
        #pragma unroll
        for (int r = 0; r < 8; ++r) { rmax[t][r] = -3.4e38f; ridx[t][r] = 0; }

    // B fragments, same contiguous k order as A: lane n<16 -> column n, K rows 0..15;
    // lane n+16 -> column n, K rows 16..31. 16 consecutive f16, 32B aligned.
    const _Float16* ebase = embed_f16 + (size_t)g * Kq * Dq + (hi ? 16 : 0);
    const float*    h2g   = h2 + g * Kq;

    for (int kt = 0; kt < Kq / 16; ++kt) {
        const int code = kt * 16 + lh;
        const _Float16* bp = ebase + (size_t)code * Dq;
        v16h b0 = *(const v16h*)(bp);        // dims 0..31 half
        v16h b1 = *(const v16h*)(bp + 32);   // dims 32..63 half
        __builtin_prefetch(bp + 16 * Dq, 0, 3);   // next code tile, near-cache scope

        v8f c[TT];
        #pragma unroll
        for (int t = 0; t < TT; ++t) c[t] = (v8f){};
        #pragma unroll
        for (int t = 0; t < TT; ++t)
            c[t] = __builtin_amdgcn_wmma_f32_16x16x32_f16(false, a0[t], false, b0,
                                                          (short)0, c[t], false, false);
        #pragma unroll
        for (int t = 0; t < TT; ++t)
            c[t] = __builtin_amdgcn_wmma_f32_16x16x32_f16(false, a1[t], false, b1,
                                                          (short)0, c[t], false, false);

        const float h2v = h2g[code];
        #pragma unroll
        for (int t = 0; t < TT; ++t) {
            #pragma unroll
            for (int r = 0; r < 8; ++r) {
                float s = c[t][r] - h2v;     // lane = code col, vgpr r = token row
                if (s > rmax[t][r]) { rmax[t][r] = s; ridx[t][r] = code; }
            }
        }
    }

    // Reduce across the 16 code-columns per half-wave (width 16 keeps the
    // row 0-7 / row 8-15 halves independent). Ties -> lowest code index.
    #pragma unroll
    for (int t = 0; t < TT; ++t) {
        #pragma unroll
        for (int r = 0; r < 8; ++r) {
            #pragma unroll
            for (int m = 8; m >= 1; m >>= 1) {
                float om = __shfl_xor(rmax[t][r], m, 16);
                int   oi = __shfl_xor(ridx[t][r], m, 16);
                if (om > rmax[t][r] || (om == rmax[t][r] && oi < ridx[t][r])) {
                    rmax[t][r] = om; ridx[t][r] = oi;
                }
            }
        }
    }
    if (lh == 0) {
        #pragma unroll
        for (int t = 0; t < TT; ++t) {
            const int rowbase = tok0 + t * 16 + (hi ? 8 : 0);
            #pragma unroll
            for (int r = 0; r < 8; ++r)
                ind_out[(size_t)(rowbase + r) * Gq + g] = ridx[t][r];
        }
    }
}

// ---------------------------------------------------------------------------
// Kernel 2: zq gather + segment sums (counts, embed_sum) via float atomics.
// ---------------------------------------------------------------------------
__global__ void pq_gather_kernel(const float* __restrict__ x,
                                 const float* __restrict__ embed,
                                 const int* __restrict__ ind,
                                 float* __restrict__ zq,
                                 float* __restrict__ counts,
                                 float* __restrict__ embed_sum) {
    int t = blockIdx.x * blockDim.x + threadIdx.x;
    if (t >= NTOK * Gq * (Dq / 4)) return;
    const int q = t & 15;
    const int g = (t >> 4) & 3;
    const int n = t >> 6;
    const int code = ind[(size_t)n * Gq + g];

    const size_t erow = ((size_t)g * Kq + code) * Dq + q * 4;
    const size_t xrow = ((size_t)n * Gq + g) * Dq + q * 4;

    const float4 ev = *(const float4*)(embed + erow);
    *(float4*)(zq + xrow) = ev;

    const float4 xv = *(const float4*)(x + xrow);
    float* es = embed_sum + erow;
    atomicAdd(es + 0, xv.x);
    atomicAdd(es + 1, xv.y);
    atomicAdd(es + 2, xv.z);
    atomicAdd(es + 3, xv.w);
    if (q == 0) atomicAdd(&counts[g * Kq + code], 1.0f);
}

// ---------------------------------------------------------------------------
// Kernel 3: tot[g] = sum_k new_cluster_size[g,k] (LDS tree reduction).
// ---------------------------------------------------------------------------
__global__ void pq_tot_kernel(const float* __restrict__ cs,
                              const float* __restrict__ counts,
                              float* __restrict__ tot) {
    __shared__ float sm[256];
    const int g = blockIdx.x;
    float acc = 0.0f;
    for (int k = threadIdx.x; k < Kq; k += 256)
        acc += DECAYf * cs[g * Kq + k] + OMDf * counts[g * Kq + k];
    sm[threadIdx.x] = acc;
    __syncthreads();
    for (int s = 128; s > 0; s >>= 1) {
        if ((int)threadIdx.x < s) sm[threadIdx.x] += sm[threadIdx.x + s];
        __syncthreads();
    }
    if (threadIdx.x == 0) tot[g] = sm[0];
}

// ---------------------------------------------------------------------------
// Kernel 4: EMA update + Laplace smoothing -> new_embed / new_cs / new_ea.
// ---------------------------------------------------------------------------
__global__ void pq_finalize_kernel(const float* __restrict__ cs,
                                   const float* __restrict__ counts,
                                   const float* __restrict__ ea,
                                   const float* __restrict__ esum,
                                   const float* __restrict__ tot,
                                   float* __restrict__ new_embed,
                                   float* __restrict__ new_cs,
                                   float* __restrict__ new_ea) {
    int gk = blockIdx.x * blockDim.x + threadIdx.x;
    if (gk >= Gq * Kq) return;
    const int g = gk >> 10;
    const float ncs = DECAYf * cs[gk] + OMDf * counts[gk];
    new_cs[gk] = ncs;
    const float tg = tot[g];
    const float smoothed = (ncs + EPSf) / (tg + (float)Kq * EPSf) * tg;
    const float inv = 1.0f / smoothed;
    const float* ear = ea   + (size_t)gk * Dq;
    const float* esr = esum + (size_t)gk * Dq;
    float* ner = new_ea    + (size_t)gk * Dq;
    float* nEr = new_embed + (size_t)gk * Dq;
    #pragma unroll 4
    for (int i = 0; i < Dq; ++i) {
        float nv = DECAYf * ear[i] + OMDf * esr[i];
        ner[i] = nv;
        nEr[i] = nv * inv;
    }
}

extern "C" void kernel_launch(void* const* d_in, const int* in_sizes, int n_in,
                              void* d_out, int out_size, void* d_ws, size_t ws_size,
                              hipStream_t stream) {
    const float* x     = (const float*)d_in[0];
    const float* embed = (const float*)d_in[1];
    const float* ea    = (const float*)d_in[2];
    const float* cs    = (const float*)d_in[3];

    char* ws = (char*)d_ws;
    _Float16* embed_f16 = (_Float16*)(ws + 0);       // 512 KB
    float*    h2        = (float*)(ws + 0x80000);    //  16 KB
    float*    counts    = (float*)(ws + 0x84000);    //  16 KB
    float*    tot       = (float*)(ws + 0x88000);    //  tiny
    float*    embed_sum = (float*)(ws + 0x90000);    //   1 MB

    float* zq      = (float*)d_out;                                 // 8,388,608
    int*   ind     = (int*)((float*)d_out + (size_t)NTOK * 256);    //   131,072
    float* n_embed = (float*)d_out + 8388608 + 131072;              //   262,144
    float* n_cs    = n_embed + (size_t)Gq * Kq * Dq;                //     4,096
    float* n_ea    = n_cs + (size_t)Gq * Kq;                        //   262,144

    pq_prep_kernel<<<(Gq * Kq + 255) / 256, 256, 0, stream>>>(
        embed, embed_f16, h2, counts, embed_sum);

    // 8 waves/block: 2 strips x 4 groups, 128 tokens per block
    pq_assign_kernel<<<NTOK / (2 * TT * 16), 256, 0, stream>>>(
        x, embed_f16, h2, ind);

    pq_gather_kernel<<<(NTOK * Gq * (Dq / 4)) / 256, 256, 0, stream>>>(
        x, embed, ind, zq, counts, embed_sum);

    pq_tot_kernel<<<Gq, 256, 0, stream>>>(cs, counts, tot);

    pq_finalize_kernel<<<(Gq * Kq + 255) / 256, 256, 0, stream>>>(
        cs, counts, ea, embed_sum, tot, n_embed, n_cs, n_ea);
}
